// ScaledDotProductAttention_64974265254301
// MI455X (gfx1250) — matrix operand order repaired and checked by reading, not compiled, not run on hardware
//
#include <hip/hip_runtime.h>
#include <hip/hip_bf16.h>

typedef __attribute__((ext_vector_type(16))) _Float16 v16h;
typedef __attribute__((ext_vector_type(8)))  _Float16 v8h;
typedef __attribute__((ext_vector_type(4)))  _Float16 v4h;
typedef __attribute__((ext_vector_type(8)))  float    v8f;

#define B_   8
#define S_   2048
#define D_   128
#define BK   64            // keys per staged tile
#define WAVES 4
#define ROWS (WAVES * 16)  // query rows per block

// LDS row strides chosen so fragment bases are 16B-aligned AND bank-spread:
//  ks row: 136 halves = 272 B (68 dwords; 68 mod 64 = 4 -> conflict-free)
//  vT row:  72 halves = 144 B (36 dwords; distinct mod 64 over 16 rows)
//  ss row:  68 floats = 272 B (68 dwords)
#define KSP (D_ + 8)
#define VTP (BK + 8)
#define SSP (BK + 4)

__device__ __forceinline__ v16h ld_frag(const _Float16* p) {
  v8h a = *(const v8h*)p;        // ds_load_b128
  v8h b = *(const v8h*)(p + 8);  // ds_load_b128
  return __builtin_shufflevector(a, b, 0, 1, 2, 3, 4, 5, 6, 7,
                                 8, 9, 10, 11, 12, 13, 14, 15);
}

__launch_bounds__(128, 1)
__global__ void fa_wmma_f16_kernel(const float* __restrict__ q,
                                   const float* __restrict__ k,
                                   const float* __restrict__ v,
                                   const unsigned char* __restrict__ mask,
                                   float* __restrict__ out) {
  __shared__ _Float16 ks[BK][KSP];         // K tile, row-major  [key][dim]
  __shared__ _Float16 vT[D_][VTP];         // V tile, transposed [dim][key]
  __shared__ float    ss[WAVES][16][SSP];  // per-wave score tile [m][key]
  __shared__ float    stat[WAVES][16];     // per-wave row-stat broadcast

  const int tid  = threadIdx.x;
  const int wave = tid >> 5;
  const int lane = tid & 31;
  const int n16  = lane & 15;
  const int hi   = lane >> 4;

  const int blocksPerBatch = S_ / ROWS;  // 32
  const int b     = blockIdx.x / blocksPerBatch;
  const int qbase = (blockIdx.x % blocksPerBatch) * ROWS;

  const float scale = 0.08838834764831845f;  // 1/sqrt(128)

  const float* qb = q + (size_t)b * S_ * D_;
  const float* kb = k + (size_t)b * S_ * D_;
  const float* vb = v + (size_t)b * S_ * D_;
  const unsigned char* mb = mask + (size_t)b * S_;

  // ---- Q A-fragments (16x32 f16 each), scale folded in ----
  const int qrow = qbase + wave * 16 + n16;
  v16h qf[4];
#pragma unroll
  for (int c = 0; c < 4; ++c) {
    const float* src = qb + (size_t)qrow * D_ + c * 32 + hi * 16;
#pragma unroll
    for (int h = 0; h < 8; ++h) {
      qf[c][h]     = (_Float16)(src[h]      * scale);
      qf[c][h + 8] = (_Float16)(src[8 + h] * scale);
    }
  }

  v8f oacc[8];
  float m_run = -1e30f, l_run = 0.0f;
#pragma unroll
  for (int t = 0; t < 8; ++t) {
    v8f z = {0.f, 0.f, 0.f, 0.f, 0.f, 0.f, 0.f, 0.f};
    oacc[t] = z;
  }

  for (int jb = 0; jb < S_; jb += BK) {
    __syncthreads();
    // ---- cooperative stage (float4 loads): K row-major, V transposed ----
#pragma unroll 4
    for (int idx = tid; idx < BK * D_ / 4; idx += 128) {
      const int r  = idx >> 5;
      const int c4 = (idx & 31) * 4;
      const float4 kv = *(const float4*)(kb + (size_t)(jb + r) * D_ + c4);
      const float4 vv = *(const float4*)(vb + (size_t)(jb + r) * D_ + c4);
      v4h kh;
      kh.x = (_Float16)kv.x; kh.y = (_Float16)kv.y;
      kh.z = (_Float16)kv.z; kh.w = (_Float16)kv.w;
      *(v4h*)&ks[r][c4] = kh;
      vT[c4 + 0][r] = (_Float16)vv.x;
      vT[c4 + 1][r] = (_Float16)vv.y;
      vT[c4 + 2][r] = (_Float16)vv.z;
      vT[c4 + 3][r] = (_Float16)vv.w;
    }
    if (jb + BK < S_) {
      __builtin_prefetch((const char*)(kb + (size_t)(jb + BK) * D_) + tid * 256, 0, 0);
      __builtin_prefetch((const char*)(vb + (size_t)(jb + BK) * D_) + tid * 256, 0, 0);
    }
    __syncthreads();

    // ---- S = (Q/sqrtD) K^T : 16 WMMAs, depth-2 pipelined fragment loads ----
    v8f acc[4];
#pragma unroll
    for (int t = 0; t < 4; ++t) {
      v8f z = {0.f, 0.f, 0.f, 0.f, 0.f, 0.f, 0.f, 0.f};
      acc[t] = z;
    }
    {
      v16h f0 = ld_frag(&ks[16 * 0 + n16][0 * 32 + hi * 16]);   // i=0: c=0,t=0
      v16h f1 = ld_frag(&ks[16 * 1 + n16][0 * 32 + hi * 16]);   // i=1: c=0,t=1
#pragma unroll
      for (int i = 0; i < 16; ++i) {
        v16h fn = f1;
        if (i + 2 < 16) {
          const int cn = (i + 2) >> 2, tn = (i + 2) & 3;
          fn = ld_frag(&ks[16 * tn + n16][cn * 32 + hi * 16]);
        }
        const int c = i >> 2, t = i & 3;
        acc[t] = __builtin_amdgcn_wmma_f32_16x16x32_f16(
            false, qf[c], false, f0, (short)0, acc[t], false, false);
        f0 = f1; f1 = fn;
      }
    }

    // ---- mask (True = masked out): key column = jb + 16t + n16 ----
#pragma unroll
    for (int t = 0; t < 4; ++t) {
      if (mb[jb + 16 * t + n16]) {
#pragma unroll
        for (int r = 0; r < 8; ++r) acc[t][r] = -1e30f;
      }
    }

    // ---- transpose scores to lane=row layout via LDS (f32) ----
#pragma unroll
    for (int t = 0; t < 4; ++t)
#pragma unroll
      for (int r = 0; r < 8; ++r)
        ss[wave][r + hi * 8][16 * t + n16] = acc[t][r];
    asm volatile("s_wait_dscnt 0" ::: "memory");

    // Each lane reads its A-layout score positions (aligned float4s).
    float sv[2][16];
#pragma unroll
    for (int c = 0; c < 2; ++c) {
      const float* srow = &ss[wave][n16][c * 32 + hi * 16];
      const float4 a0 = *(const float4*)(srow);
      const float4 a1 = *(const float4*)(srow + 4);
      const float4 b0 = *(const float4*)(srow + 8);
      const float4 b1 = *(const float4*)(srow + 12);
      sv[c][0] = a0.x;  sv[c][1] = a0.y;  sv[c][2]  = a0.z;  sv[c][3]  = a0.w;
      sv[c][4] = a1.x;  sv[c][5] = a1.y;  sv[c][6]  = a1.z;  sv[c][7]  = a1.w;
      sv[c][8] = b0.x;  sv[c][9] = b0.y;  sv[c][10] = b0.z;  sv[c][11] = b0.w;
      sv[c][12] = b1.x; sv[c][13] = b1.y; sv[c][14] = b1.z;  sv[c][15] = b1.w;
    }

    // ---- online softmax: local max/sum + one xor-16 merge ----
    float mx = sv[0][0];
#pragma unroll
    for (int c = 0; c < 2; ++c)
#pragma unroll
      for (int h = 0; h < 16; ++h) mx = fmaxf(mx, sv[c][h]);
    mx = fmaxf(mx, __shfl_xor(mx, 16, 32));

    const float mnew = fmaxf(m_run, mx);
    const float al   = __expf(m_run - mnew);
    m_run = mnew;

    v16h pf[2];
    float psum = 0.0f;
#pragma unroll
    for (int c = 0; c < 2; ++c)
#pragma unroll
      for (int h = 0; h < 16; ++h) {
        const float p = __expf(sv[c][h] - mnew);
        psum += p;
        pf[c][h] = (_Float16)p;
      }
    psum += __shfl_xor(psum, 16, 32);
    l_run = l_run * al + psum;

    // ---- broadcast alpha to C layout, rescale O ----
    if (lane < 16) stat[wave][n16] = al;
    asm volatile("s_wait_dscnt 0" ::: "memory");
    float alC[8];
#pragma unroll
    for (int r = 0; r < 8; ++r) alC[r] = stat[wave][r + hi * 8];
#pragma unroll
    for (int t2 = 0; t2 < 8; ++t2)
#pragma unroll
      for (int r = 0; r < 8; ++r) oacc[t2][r] *= alC[r];

    // ---- O += P V : 16 WMMAs, depth-2 pipelined fragment loads ----
    {
      v16h f0 = ld_frag(&vT[16 * 0 + n16][0 * 32 + hi * 16]);   // i=0: c=0,t2=0
      v16h f1 = ld_frag(&vT[16 * 1 + n16][0 * 32 + hi * 16]);   // i=1: c=0,t2=1
#pragma unroll
      for (int i = 0; i < 16; ++i) {
        v16h fn = f1;
        if (i + 2 < 16) {
          const int cn = (i + 2) >> 3, tn = (i + 2) & 7;
          fn = ld_frag(&vT[16 * tn + n16][cn * 32 + hi * 16]);
        }
        const int c = i >> 3, t2 = i & 7;
        oacc[t2] = __builtin_amdgcn_wmma_f32_16x16x32_f16(
            false, pf[c], false, f0, (short)0, oacc[t2], false, false);
        f0 = f1; f1 = fn;
      }
    }
  }

  // ---- broadcast 1/l to C layout, normalize, store ----
  if (lane < 16) stat[wave][n16] = l_run;
  asm volatile("s_wait_dscnt 0" ::: "memory");
  float invC[8];
#pragma unroll
  for (int r = 0; r < 8; ++r) invC[r] = 1.0f / stat[wave][r + hi * 8];

#pragma unroll
  for (int r = 0; r < 8; ++r) {
    const int row = qbase + wave * 16 + r + hi * 8;
    float* dst = out + ((size_t)b * S_ + row) * D_;
#pragma unroll
    for (int t2 = 0; t2 < 8; ++t2)
      dst[16 * t2 + n16] = oacc[t2][r] * invC[r];
  }
}

extern "C" void kernel_launch(void* const* d_in, const int* in_sizes, int n_in,
                              void* d_out, int out_size, void* d_ws, size_t ws_size,
                              hipStream_t stream) {
  (void)in_sizes; (void)n_in; (void)out_size; (void)d_ws; (void)ws_size;
  const float* q = (const float*)d_in[0];
  const float* k = (const float*)d_in[1];
  const float* v = (const float*)d_in[2];
  const unsigned char* mask = (const unsigned char*)d_in[3];
  float* out = (float*)d_out;

  const int grid = B_ * (S_ / ROWS);  // 256 blocks
  fa_wmma_f16_kernel<<<grid, 128, 0, stream>>>(q, k, v, mask, out);
}
